// NnHalfKA_49555332661321
// MI455X (gfx1250) — matrix operand order repaired and checked by reading, not compiled, not run on hardware
//
#include <hip/hip_runtime.h>
#include <math.h>

typedef __attribute__((ext_vector_type(16))) _Float16 v16h;
typedef __attribute__((ext_vector_type(8)))  _Float16 v8h;
typedef __attribute__((ext_vector_type(4)))  _Float16 v4h;
typedef __attribute__((ext_vector_type(8)))  float    v8f;

#define FT_IN   49152
#define VFT_IN  768
#define FT_OUT  512
#define BATCH   8192
#define MAX_F   32
#define NNZ     (BATCH * MAX_F)
#define ROWS_PB 16                 // batch rows per block
#define LDS_STR 1032               // 1024 hidden cols + 8 halfs pad (kills LDS bank conflicts)

__device__ __forceinline__ void fma4(float4& acc, float v, const float4& a, const float4& c) {
    acc.x = fmaf(v, a.x + c.x, acc.x);
    acc.y = fmaf(v, a.y + c.y, acc.y);
    acc.z = fmaf(v, a.z + c.z, acc.z);
    acc.w = fmaf(v, a.w + c.w, acc.w);
}

__device__ __forceinline__ float clamp01(float x) { return fminf(fmaxf(x, 0.0f), 1.0f); }

__global__ __launch_bounds__(1024)
void nnue_halfka_fused(const int*   __restrict__ stm_idx,     // [2, NNZ] flat
                       const int*   __restrict__ nstm_idx,    // [2, NNZ] flat
                       const float* __restrict__ values,      // [NNZ]
                       const float* __restrict__ W_ft,        // [49152, 512]
                       const float* __restrict__ b_ft,        // [512]
                       const float* __restrict__ W_fft,       // [768, 512]
                       const float* __restrict__ b_fft,       // [512]
                       const float* __restrict__ W_out,       // [1024, 1]
                       const float* __restrict__ b_out,       // [1]
                       float*       __restrict__ out)         // [8192]
{
    __shared__ __align__(32) _Float16 lds_h[ROWS_PB][LDS_STR];   // clamped hidden, f16
    __shared__ __align__(32) _Float16 lds_wout[2 * FT_OUT];      // W_out as f16

    const int tid  = threadIdx.x;
    const int lane = tid & 31;
    const int wid  = tid >> 5;

    // Stage W_out (1024 f32 -> f16) into LDS once per block.
    if (tid < 2 * FT_OUT) lds_wout[tid] = (_Float16)W_out[tid];

    // ---------------- Phase 1: sparse feature transformer ----------------
    // wave -> (row-in-block, perspective); each lane owns 16 output floats (4 x float4).
    const int r = wid >> 1;                 // 0..15
    const int p = wid & 1;                  // 0 = stm, 1 = nstm
    const int b = blockIdx.x * ROWS_PB + r; // batch row

    const int* __restrict__ ind = p ? nstm_idx : stm_idx;

    // Features of row b are contiguous: lane l pre-loads feature l, broadcast via shfl.
    const unsigned my_feat = (unsigned)ind[NNZ + b * MAX_F + lane];
    const float    my_val  = values[b * MAX_F + lane];

    float4 acc0 = {0,0,0,0}, acc1 = {0,0,0,0}, acc2 = {0,0,0,0}, acc3 = {0,0,0,0};

    const float4* __restrict__ Wft4  = (const float4*)W_ft;    // row stride 128 float4
    const float4* __restrict__ Wfft4 = (const float4*)W_fft;

    #pragma unroll 4
    for (int i = 0; i < MAX_F; ++i) {
        const unsigned feat = (unsigned)__shfl((int)my_feat, i, 32);
        const float    v    = __shfl(my_val, i, 32);

        // Prefetch next feature's W_ft row (L2-resident gather; hides latency).
        if (i + 1 < MAX_F) {
            const unsigned nf = (unsigned)__shfl((int)my_feat, i + 1, 32);
            __builtin_prefetch(&W_ft[(size_t)nf * FT_OUT + (size_t)lane * 16], 0, 3);
        }

        const size_t fb  = (size_t)feat * (FT_OUT / 4);
        const size_t ffb = (size_t)(feat % VFT_IN) * (FT_OUT / 4);

        const float4 a0 = Wft4[fb + lane];        const float4 c0 = Wfft4[ffb + lane];
        const float4 a1 = Wft4[fb + 32 + lane];   const float4 c1 = Wfft4[ffb + 32 + lane];
        const float4 a2 = Wft4[fb + 64 + lane];   const float4 c2 = Wfft4[ffb + 64 + lane];
        const float4 a3 = Wft4[fb + 96 + lane];   const float4 c3 = Wfft4[ffb + 96 + lane];

        fma4(acc0, v, a0, c0);
        fma4(acc1, v, a1, c1);
        fma4(acc2, v, a2, c2);
        fma4(acc3, v, a3, c3);
    }

    // Bias + clamp, convert to f16, stash in LDS (row-major, padded stride).
    {
        const float4* __restrict__ bft4  = (const float4*)b_ft;
        const float4* __restrict__ bfft4 = (const float4*)b_fft;
        float4 accs[4] = {acc0, acc1, acc2, acc3};
        #pragma unroll
        for (int seg = 0; seg < 4; ++seg) {
            const int cidx = seg * 32 + lane;            // float4 index within 512/4
            const float4 ba = bft4[cidx];
            const float4 bb = bfft4[cidx];
            v4h hv;
            hv[0] = (_Float16)clamp01(accs[seg].x + ba.x + bb.x);
            hv[1] = (_Float16)clamp01(accs[seg].y + ba.y + bb.y);
            hv[2] = (_Float16)clamp01(accs[seg].z + ba.z + bb.z);
            hv[3] = (_Float16)clamp01(accs[seg].w + ba.w + bb.w);
            *(v4h*)&lds_h[r][p * FT_OUT + cidx * 4] = hv;
        }
    }

    __syncthreads();

    // ---------------- Phase 2: output GEMV via WMMA (wave 0 only) ----------------
    // D[16,16] += A[16,32] x B[32,16], A = hidden tile (f16), B = W_out broadcast
    // across N, so every column of D holds the full 1024-long dot product.
    if (wid == 0) {
        const int m    = lane & 15;              // A: lanes 0-15 and 16-31 both cover M=0..15
        const int ksel = (lane < 16) ? 0 : 16;   // A layout: lane<16 -> K 0-15, lane>=16 -> K 16-31
        const int bsel = (lane < 16) ? 0 : 16;   // B, same contiguous order as A: lane<16 -> K rows 0-15

        v8f c = {};
        #pragma unroll 8
        for (int chunk = 0; chunk < 32; ++chunk) {
            const int k0 = chunk * 32;
            const v8h lo = *(const v8h*)&lds_h[m][k0 + ksel];
            const v8h hi = *(const v8h*)&lds_h[m][k0 + 8 + ksel];
            v16h a;
            #pragma unroll
            for (int j = 0; j < 8; ++j) { a[j] = lo[j]; a[8 + j] = hi[j]; }
            const v16h bvec = *(const v16h*)&lds_wout[k0 + bsel];
            c = __builtin_amdgcn_wmma_f32_16x16x32_f16(
                    false, a, false, bvec, (short)0, c, false, false);
        }

        const float bo = b_out[0];
        const int   b0 = blockIdx.x * ROWS_PB;
        // C/D layout: VGPR v, lane 0 -> (M=v, N=0); lane 16 -> (M=v+8, N=0).
        if (lane == 0) {
            #pragma unroll
            for (int v = 0; v < 8; ++v)
                out[b0 + v] = 1.0f / (1.0f + expf(-(c[v] + bo)));
        } else if (lane == 16) {
            #pragma unroll
            for (int v = 0; v < 8; ++v)
                out[b0 + 8 + v] = 1.0f / (1.0f + expf(-(c[v] + bo)));
        }
    }
}

extern "C" void kernel_launch(void* const* d_in, const int* in_sizes, int n_in,
                              void* d_out, int out_size, void* d_ws, size_t ws_size,
                              hipStream_t stream) {
    (void)in_sizes; (void)n_in; (void)d_ws; (void)ws_size; (void)out_size;
    const int*   stm_idx  = (const int*)  d_in[0];
    const int*   nstm_idx = (const int*)  d_in[1];
    const float* values   = (const float*)d_in[2];
    // d_in[3] = size scalar (8192, compile-time constant here)
    const float* W_ft     = (const float*)d_in[4];
    const float* b_ft     = (const float*)d_in[5];
    const float* W_fft    = (const float*)d_in[6];
    const float* b_fft    = (const float*)d_in[7];
    const float* W_out    = (const float*)d_in[8];
    const float* b_out    = (const float*)d_in[9];
    float*       out      = (float*)d_out;

    dim3 grid(BATCH / ROWS_PB);   // 512 blocks
    dim3 block(1024);             // 32 waves: 16 rows x 2 perspectives
    nnue_halfka_fused<<<grid, block, 0, stream>>>(
        stm_idx, nstm_idx, values, W_ft, b_ft, W_fft, b_fft, W_out, b_out, out);
}
